// RNN_57346403336386
// MI455X (gfx1250) — matrix operand order repaired and checked by reading, not compiled, not run on hardware
//
#include <hip/hip_runtime.h>
#include <math.h>

// ---------------- problem constants ----------------
#define BATCH 2048
#define SEQ_T 80
#define EMB_D 512
#define UNITS 1024

// ---------------- GEMM tiling ----------------
#define BM 128
#define BN 128
#define BK 32
#define LDSTRIDE (BK + 8)   // 40 elems = 80B row stride (multiple of 16B)

typedef __attribute__((ext_vector_type(16))) __bf16 v16bf;
typedef __attribute__((ext_vector_type(8)))  float  v8f;

union Frag { v16bf v; uint4 u[2]; };

// -------- CDNA5 async global->LDS copy (ASYNCcnt-tracked), with fallback --------
#if defined(__has_builtin)
#  if __has_builtin(__builtin_amdgcn_global_load_async_to_lds_b128)
#    define USE_ASYNC_LDS 1
#  endif
#  if __has_builtin(__builtin_amdgcn_s_wait_asynccnt)
#    define HAVE_WAIT_ASYNC 1
#  endif
#endif
#ifndef USE_ASYNC_LDS
#  define USE_ASYNC_LDS 0
#endif
#ifndef HAVE_WAIT_ASYNC
#  define HAVE_WAIT_ASYNC 0
#endif

// parameter pointee type per hipcc diagnostic: '__vector_size__(4*sizeof(int)) int'
typedef int async_v4i __attribute__((__vector_size__(4 * sizeof(int))));
typedef __attribute__((address_space(1))) async_v4i* g_v4i_ptr;
typedef __attribute__((address_space(3))) async_v4i* l_v4i_ptr;

__device__ __forceinline__ void cp16B_g2l(const __bf16* g, __bf16* l) {
#if USE_ASYNC_LDS
    __builtin_amdgcn_global_load_async_to_lds_b128(
        (g_v4i_ptr)(void*)g, (l_v4i_ptr)l, 0, 0);
#else
    *(uint4*)l = *(const uint4*)g;
#endif
}

__device__ __forceinline__ void wait_async_copies() {
#if USE_ASYNC_LDS
#  if HAVE_WAIT_ASYNC
    __builtin_amdgcn_s_wait_asynccnt(0);
#  else
    asm volatile("s_wait_asynccnt 0x0" ::: "memory");
#  endif
#endif
}

// Build concatenated, transposed bf16 weights: out[n][k] = (k<KX ? W[k][n] : U[k-KX][n])
__global__ void build_wcat(const float* __restrict__ W, const float* __restrict__ U,
                           __bf16* __restrict__ out, int KX, int KU, int N) {
    const int Ktot = KX + KU;
    size_t idx = (size_t)blockIdx.x * 256 + threadIdx.x;
    if (idx >= (size_t)N * Ktot) return;
    int n = (int)(idx / Ktot);
    int k = (int)(idx % Ktot);
    float v = (k < KX) ? W[(size_t)k * N + n] : U[(size_t)(k - KX) * N + n];
    out[idx] = (__bf16)v;
}

// One RNN step: hout = tanh( concat(X, hprev) @ Wcat^T + bias )
//   X part (k < KX):  gatherX ? bf16(emb[tokens[m][t]][k]) : xbuf[m][k] (bf16)
//   H part (k >= KX): hprev[m][k-KX] (bf16)
//   Wcat: bf16 [UNITS][KX+UNITS]  (N-major, K contiguous)
__global__ __launch_bounds__(256)
void rnn_step_gemm(const int* __restrict__ tokens, const float* __restrict__ emb,
                   int t, int gatherX,
                   const __bf16* __restrict__ xbuf,
                   const __bf16* __restrict__ hprev,
                   const __bf16* __restrict__ wcat,
                   const float* __restrict__ bias,
                   __bf16* __restrict__ hout,
                   int KX) {
    const int Ktot = KX + UNITS;

    __shared__ __align__(16) __bf16 As[2][BM][LDSTRIDE];
    __shared__ __align__(16) __bf16 Bs[2][BN][LDSTRIDE];

    const int tid    = threadIdx.x;
    const int blockM = blockIdx.y * BM;
    const int blockN = blockIdx.x * BN;

    // wave geometry: 8 waves, 4 (M) x 2 (N); each wave owns a 32x64 output tile
    const int w    = tid >> 5;
    const int lane = tid & 31;
    const int wm   = w >> 1;        // 0..3
    const int wn   = w & 1;         // 0..1
    const int lm   = lane & 15;
    const int lhi  = lane >> 4;

    // loader geometry: each thread owns one 16-element (32B) chunk of one row
    const int lrow  = tid >> 1;     // 0..127
    const int lhalf = tid & 1;      // 0 or 1
    const int am    = blockM + lrow;
    const int bnrow = blockN + lrow;

    int tok = 0;
    if (gatherX) tok = tokens[am * SEQ_T + t];

    // issue one pipeline stage: copy A & B 32B chunks for K-chunk kk into buffer `buf`
    auto issue = [&](int kk, int buf) {
        const int kc = kk + lhalf * 16;
        __bf16* adst = &As[buf][lrow][lhalf * 16];
        if (kc < KX) {              // KX % 32 == 0 -> never wave-divergent
            if (gatherX) {
                const float* src = emb + (size_t)tok * EMB_D + kc;
                union { __bf16 h[16]; uint4 u[2]; } pk;
#pragma unroll
                for (int q = 0; q < 4; ++q) {
                    float4 f = ((const float4*)src)[q];
                    pk.h[q * 4 + 0] = (__bf16)f.x;
                    pk.h[q * 4 + 1] = (__bf16)f.y;
                    pk.h[q * 4 + 2] = (__bf16)f.z;
                    pk.h[q * 4 + 3] = (__bf16)f.w;
                }
                ((uint4*)adst)[0]   = pk.u[0];
                *(uint4*)(adst + 8) = pk.u[1];
            } else {
                const __bf16* s = xbuf + (size_t)am * KX + kc;
                cp16B_g2l(s,     adst);
                cp16B_g2l(s + 8, adst + 8);
            }
        } else {
            const __bf16* s = hprev + (size_t)am * UNITS + (kc - KX);
            cp16B_g2l(s,     adst);
            cp16B_g2l(s + 8, adst + 8);
        }
        const __bf16* wsrc = wcat + (size_t)bnrow * Ktot + kc;
        __bf16* bdst = &Bs[buf][lrow][lhalf * 16];
        cp16B_g2l(wsrc,     bdst);
        cp16B_g2l(wsrc + 8, bdst + 8);
    };

    v8f acc[2][4];
#pragma unroll
    for (int i = 0; i < 2; ++i)
#pragma unroll
        for (int j = 0; j < 4; ++j)
#pragma unroll
            for (int r = 0; r < 8; ++r) acc[i][j][r] = 0.f;

    issue(0, 0);   // prologue

    for (int kk = 0; kk < Ktot; kk += BK) {
        const int buf = (kk >> 5) & 1;

        wait_async_copies();     // my stage-n async copies have landed in LDS
        __syncthreads();         // everyone's stage-n data ready; stage n-1 reads done

        if (kk + BK < Ktot) issue(kk + BK, buf ^ 1);   // overlap copy with compute

        // ---- fragments ----
        // A 16x32 bf16: lane<16 holds M=lm, K 0..15; lane>=16 holds K 16..31
        Frag fa[2];
#pragma unroll
        for (int i = 0; i < 2; ++i) {
            const __bf16* ap = &As[buf][wm * 32 + i * 16 + lm][lhi * 16];
            fa[i].u[0] = *(const uint4*)ap;
            fa[i].u[1] = *(const uint4*)(ap + 8);
        }
        // B 32x16 bf16: lane<16 holds N=lm, K 0..15; lane>=16 holds K 16..31
        Frag fb[4];
#pragma unroll
        for (int j = 0; j < 4; ++j) {
            const __bf16* bp = &Bs[buf][wn * 64 + j * 16 + lm][lhi * 16];
            fb[j].u[0] = *(const uint4*)bp;
            fb[j].u[1] = *(const uint4*)(bp + 8);
        }

#pragma unroll
        for (int i = 0; i < 2; ++i)
#pragma unroll
            for (int j = 0; j < 4; ++j)
                acc[i][j] = __builtin_amdgcn_wmma_f32_16x16x32_bf16(
                    false, fa[i].v, false, fb[j].v, (short)0, acc[i][j], false, false);
    }

    // ---- epilogue: + bias, tanh, store bf16 ----
    // C layout: VGPR r, lanes 0-15 -> M=r, N=lane; lanes 16-31 -> M=8+r, N=lane-16
#pragma unroll
    for (int i = 0; i < 2; ++i) {
#pragma unroll
        for (int j = 0; j < 4; ++j) {
            const int mbase = blockM + wm * 32 + i * 16 + (lhi << 3);
            const int n     = blockN + wn * 64 + j * 16 + lm;
            const float bb  = bias[n];
#pragma unroll
            for (int r = 0; r < 8; ++r) {
                float vv = acc[i][j][r] + bb;
                hout[(size_t)(mbase + r) * UNITS + n] = (__bf16)tanhf(vv);
            }
        }
    }
}

// logits = sigmoid(h2 @ Wo + bo), one block per batch row
__global__ __launch_bounds__(256)
void out_proj(const __bf16* __restrict__ h2, const float* __restrict__ Wo,
              const float* __restrict__ bo, float* __restrict__ out) {
    __shared__ float red[256];
    const int b = blockIdx.x;
    float s = 0.f;
    for (int k = threadIdx.x; k < UNITS; k += 256)
        s += (float)h2[(size_t)b * UNITS + k] * Wo[k];
    red[threadIdx.x] = s;
    __syncthreads();
    for (int wdt = 128; wdt > 0; wdt >>= 1) {
        if (threadIdx.x < wdt) red[threadIdx.x] += red[threadIdx.x + wdt];
        __syncthreads();
    }
    if (threadIdx.x == 0) {
        float l = red[0] + bo[0];
        out[b] = 1.f / (1.f + expf(-l));
    }
}

extern "C" void kernel_launch(void* const* d_in, const int* in_sizes, int n_in,
                              void* d_out, int out_size, void* d_ws, size_t ws_size,
                              hipStream_t stream) {
    (void)in_sizes; (void)n_in; (void)out_size; (void)ws_size;

    const int*   tokens = (const int*)d_in[0];
    const float* emb    = (const float*)d_in[1];
    const float* W1     = (const float*)d_in[2];
    const float* U1     = (const float*)d_in[3];
    const float* b1     = (const float*)d_in[4];
    const float* W2     = (const float*)d_in[5];
    const float* U2     = (const float*)d_in[6];
    const float* b2     = (const float*)d_in[7];
    const float* Wo     = (const float*)d_in[8];
    const float* bo     = (const float*)d_in[9];
    float* out = (float*)d_out;

    // workspace layout (bf16):
    //   w1cat [1024][1536] | w2cat [1024][2048] | h1 ping-pong 2x[2048][1024] | h2 ping-pong
    __bf16* ws    = (__bf16*)d_ws;
    __bf16* w1cat = ws;
    __bf16* w2cat = w1cat + (size_t)UNITS * (EMB_D + UNITS);
    __bf16* h1    = w2cat + (size_t)UNITS * (UNITS + UNITS);
    const size_t BH = (size_t)BATCH * UNITS;
    __bf16* h2    = h1 + 2 * BH;

    // h_{-1} = 0 for both layers (only the t=0 "prev" buffers need init)
    (void)hipMemsetAsync(h1, 0, BH * sizeof(__bf16), stream);
    (void)hipMemsetAsync(h2, 0, BH * sizeof(__bf16), stream);

    {
        size_t tot = (size_t)UNITS * (EMB_D + UNITS);
        build_wcat<<<(unsigned)((tot + 255) / 256), 256, 0, stream>>>(W1, U1, w1cat, EMB_D, UNITS, UNITS);
        tot = (size_t)UNITS * (UNITS + UNITS);
        build_wcat<<<(unsigned)((tot + 255) / 256), 256, 0, stream>>>(W2, U2, w2cat, UNITS, UNITS, UNITS);
    }

    dim3 grid(UNITS / BN, BATCH / BM);   // (8, 16)
    for (int t = 0; t < SEQ_T; ++t) {
        const __bf16* h1p = h1 + (size_t)(t & 1) * BH;
        __bf16*       h1c = h1 + (size_t)((t + 1) & 1) * BH;
        const __bf16* h2p = h2 + (size_t)(t & 1) * BH;
        __bf16*       h2c = h2 + (size_t)((t + 1) & 1) * BH;
        // layer 1: A = [gather(emb, tokens[:,t]) | h1_{t-1}], K = 512 + 1024
        rnn_step_gemm<<<grid, 256, 0, stream>>>(tokens, emb, t, 1,
                                                (const __bf16*)nullptr, h1p, w1cat, b1, h1c, EMB_D);
        // layer 2: A = [h1_t | h2_{t-1}], K = 1024 + 1024
        rnn_step_gemm<<<grid, 256, 0, stream>>>(nullptr, nullptr, t, 0,
                                                h1c, h2p, w2cat, b2, h2c, UNITS);
    }

    // T=80 even -> final h2 lives in buffer 0
    out_proj<<<BATCH, 256, 0, stream>>>(h2, Wo, bo, out);
}
